// ConditionalNFEncoder_13443247637227
// MI455X (gfx1250) — matrix operand order repaired and checked by reading, not compiled, not run on hardware
//
#include <hip/hip_runtime.h>

typedef __attribute__((ext_vector_type(16))) _Float16 v16h;
typedef __attribute__((ext_vector_type(8)))  _Float16 v8h;
typedef __attribute__((ext_vector_type(8)))  float    v8f;

#define Bc   32
#define Tc   2048
#define Dc   512
#define Hc   64
#define Sc   3
#define NBc  2
#define OUTC (2*Dc + 1)           // 1025
#define LDA  72                   // padded LDS row stride (halves); keeps 16B alignment
#define NW1  (Sc*NBc*Hc*Hc)       // 24576 elements per weight tensor
#define LOG_2PI 1.8378770664093453f

// ---------------------------------------------------------------------------
// Kernel 0: convert W1/W2 (f32) -> f16 in workspace, laid out identically
// ---------------------------------------------------------------------------
__global__ void cvt_weights_kernel(const float* __restrict__ W1,
                                   const float* __restrict__ W2,
                                   _Float16* __restrict__ ws) {
  int i = blockIdx.x * blockDim.x + threadIdx.x;
  if (i < NW1) {
    ws[i]       = (_Float16)W1[i];
    ws[NW1 + i] = (_Float16)W2[i];
  }
}

// ---------------------------------------------------------------------------
// Kernel 1: feature broadcast  out[b,t,0:512] = trend*Wt+bt ; [512:1024] = seasonal
// one block per (b,t); 4 fully-coalesced scalar store sweeps (1025 stride kills
// float4 alignment, so scalar b32 stores are the clean path)
// ---------------------------------------------------------------------------
__global__ void feat_kernel(const float* __restrict__ trend,
                            const float* __restrict__ seasonal,
                            const float* __restrict__ Wt, const float* __restrict__ bt,
                            const float* __restrict__ Ws, const float* __restrict__ bs,
                            float* __restrict__ out) {
  const int bt_idx = blockIdx.x;
  const int k = threadIdx.x;
  const float tv = trend[bt_idx];
  const float sv = seasonal[bt_idx];
  const size_t base = (size_t)bt_idx * OUTC;
#pragma unroll
  for (int ii = 0; ii < 4; ++ii) {
    int d = k + ii * 256;
    if (d < Dc) out[base + d] = tv * Wt[d] + bt[d];
    else        out[base + d] = sv * Ws[d - Dc] + bs[d - Dc];
  }
}

// ---------------------------------------------------------------------------
// A-fragment loader: 16x32 f16 A layout (lane = m + 16*hi):
//   halves 0..15 -> K = kc*32 + hi*16 + 0..15
// ---------------------------------------------------------------------------
__device__ __forceinline__ v16h load_a_frag(const _Float16* aT, int m, int hi, int kc) {
  const _Float16* p = aT + m * LDA + kc * 32 + hi * 16;
  v8h lo = *(const v8h*)p;
  v8h hh = *(const v8h*)(p + 8);
  return __builtin_shufflevector(lo, hh, 0,1,2,3,4,5,6,7,8,9,10,11,12,13,14,15);
}

// ---------------------------------------------------------------------------
// Kernel 2: normalizing-flow log-prob.  One wave = 16 samples.
// h kept in C/D WMMA layout; relu(h) bounced via per-wave LDS tile for the
// D->A relayout; B fragments are contiguous 16-half rows of f16 weights.
// ---------------------------------------------------------------------------
__global__ void __launch_bounds__(256)
flow_kernel(const float* __restrict__ residual,
            const float* __restrict__ b_init, const float* __restrict__ Wc0,
            const float* __restrict__ bc0,
            const float* __restrict__ b1,  const float* __restrict__ b2,
            const float* __restrict__ Wcb, const float* __restrict__ bcb,
            const float* __restrict__ Wf,  const float* __restrict__ bf,
            const _Float16* __restrict__ wW1, const _Float16* __restrict__ wW2,
            float* __restrict__ out) {
  __shared__ _Float16 smem[8 * 16 * LDA];

  const int lane = threadIdx.x & 31;
  const int wave = threadIdx.x >> 5;
  _Float16* aT = smem + wave * 16 * LDA;

  const int m16 = lane & 15;
  const int hi  = lane >> 4;
  const int gi  = (blockIdx.x * 8 + wave) * 16 + m16;   // global sample id

  const float x = residual[gi];
  const float c = ((gi & (Tc - 1)) == 0) ? 0.f : residual[gi - 1];

  // c[m] for the 8 sample-rows this lane owns in the D layout (m = r + 8*hi)
  float cm[8];
#pragma unroll
  for (int r = 0; r < 8; ++r) cm[r] = __shfl(c, r + 8 * hi, 32);

  float z = x, logdet = 0.f;

  for (int i = 0; i < Sc; ++i) {
    // ---- h = c*Wc0 + (bc0 + b_init), in D layout -------------------------
    float h[4][8];
#pragma unroll
    for (int nT = 0; nT < 4; ++nT) {
      const int n = nT * 16 + m16;
      const float wc0 = Wc0[i * Hc + n];
      const float hb  = bc0[i * Hc + n] + b_init[i * Hc + n];
#pragma unroll
      for (int r = 0; r < 8; ++r) h[nT][r] = cm[r] * wc0 + hb;
    }

#pragma unroll
    for (int j = 0; j < NBc; ++j) {
      const _Float16* w1p = wW1 + (size_t)((i * NBc + j) * Hc) * Hc;
      const _Float16* w2p = wW2 + (size_t)((i * NBc + j) * Hc) * Hc;
      float t[4][8];

      // ---- t = relu(h) @ W1^T + b1 -----------------------------------
#pragma unroll
      for (int nT = 0; nT < 4; ++nT) {
        const int n = nT * 16 + m16;
#pragma unroll
        for (int r = 0; r < 8; ++r) {
          const int m = r + 8 * hi;
          float v = h[nT][r]; v = v > 0.f ? v : 0.f;
          aT[m * LDA + n] = (_Float16)v;
        }
      }
      asm volatile("s_wait_dscnt 0" ::: "memory");
      v16h a0 = load_a_frag(aT, m16, hi, 0);
      v16h a1 = load_a_frag(aT, m16, hi, 1);
#pragma unroll
      for (int nT = 0; nT < 4; ++nT) {
        const int n = nT * 16 + m16;
        v8f acc = {};
        v16h bf0 = *(const v16h*)(w1p + n * Hc +      hi * 16);
        acc = __builtin_amdgcn_wmma_f32_16x16x32_f16(false, a0, false, bf0,
                                                     (short)0, acc, false, false);
        v16h bf1 = *(const v16h*)(w1p + n * Hc + 32 + hi * 16);
        acc = __builtin_amdgcn_wmma_f32_16x16x32_f16(false, a1, false, bf1,
                                                     (short)0, acc, false, false);
        const float bias = b1[(i * NBc + j) * Hc + n];
#pragma unroll
        for (int r = 0; r < 8; ++r) t[nT][r] = acc[r] + bias;
      }

      // ---- t2 = relu(t) @ W2^T + b2;  h += t2 * sigmoid(c*Wcb + bcb) --
#pragma unroll
      for (int nT = 0; nT < 4; ++nT) {
        const int n = nT * 16 + m16;
#pragma unroll
        for (int r = 0; r < 8; ++r) {
          const int m = r + 8 * hi;
          float v = t[nT][r]; v = v > 0.f ? v : 0.f;
          aT[m * LDA + n] = (_Float16)v;
        }
      }
      asm volatile("s_wait_dscnt 0" ::: "memory");
      a0 = load_a_frag(aT, m16, hi, 0);
      a1 = load_a_frag(aT, m16, hi, 1);
#pragma unroll
      for (int nT = 0; nT < 4; ++nT) {
        const int n = nT * 16 + m16;
        v8f acc = {};
        v16h bg0 = *(const v16h*)(w2p + n * Hc +      hi * 16);
        acc = __builtin_amdgcn_wmma_f32_16x16x32_f16(false, a0, false, bg0,
                                                     (short)0, acc, false, false);
        v16h bg1 = *(const v16h*)(w2p + n * Hc + 32 + hi * 16);
        acc = __builtin_amdgcn_wmma_f32_16x16x32_f16(false, a1, false, bg1,
                                                     (short)0, acc, false, false);
        const float bias2 = b2[(i * NBc + j) * Hc + n];
        const float wcbv  = Wcb[(i * NBc + j) * Hc + n];
        const float bcbv  = bcb[(i * NBc + j) * Hc + n];
#pragma unroll
        for (int r = 0; r < 8; ++r) {
          const float t2 = acc[r] + bias2;
          const float g  = cm[r] * wcbv + bcbv;
          const float s  = 1.f / (1.f + __expf(-g));
          h[nT][r] += t2 * s;
        }
      }
    } // j

    // ---- out = relu(h) @ Wf^T + bf (64 -> 2): VALU via LDS tile ----------
#pragma unroll
    for (int nT = 0; nT < 4; ++nT) {
      const int n = nT * 16 + m16;
#pragma unroll
      for (int r = 0; r < 8; ++r) {
        const int m = r + 8 * hi;
        float v = h[nT][r]; v = v > 0.f ? v : 0.f;
        aT[m * LDA + n] = (_Float16)v;
      }
    }
    asm volatile("s_wait_dscnt 0" ::: "memory");
    float o0 = bf[i * 2 + 0], o1 = bf[i * 2 + 1];
#pragma unroll
    for (int ch = 0; ch < 4; ++ch) {
      v16h hv = *(const v16h*)(aT + m16 * LDA + ch * 16);
#pragma unroll
      for (int k = 0; k < 16; ++k) {
        const float hk = (float)hv[k];
        o0 += hk * Wf[(i * 2 + 0) * Hc + ch * 16 + k];
        o1 += hk * Wf[(i * 2 + 1) * Hc + ch * 16 + k];
      }
    }
    const float sp    = (o0 > 20.f) ? o0 : __logf(1.f + __expf(o0));
    const float scale = sp + 1e-3f;
    z = scale * z + o1;
    logdet += __logf(scale);
  } // i

  if (lane < 16) {
    const float lp = -0.5f * z * z - 0.5f * LOG_2PI + logdet;
    out[(size_t)gi * OUTC + (OUTC - 1)] = lp;
  }
}

// ---------------------------------------------------------------------------
extern "C" void kernel_launch(void* const* d_in, const int* in_sizes, int n_in,
                              void* d_out, int out_size, void* d_ws, size_t ws_size,
                              hipStream_t stream) {
  const float* trend    = (const float*)d_in[0];
  const float* seasonal = (const float*)d_in[1];
  const float* residual = (const float*)d_in[2];
  const float* Wt       = (const float*)d_in[3];
  const float* bt       = (const float*)d_in[4];
  const float* Ws       = (const float*)d_in[5];
  const float* bs       = (const float*)d_in[6];
  const float* b_init   = (const float*)d_in[7];
  const float* Wc0      = (const float*)d_in[8];
  const float* bc0      = (const float*)d_in[9];
  const float* W1       = (const float*)d_in[10];
  const float* b1       = (const float*)d_in[11];
  const float* W2       = (const float*)d_in[12];
  const float* b2       = (const float*)d_in[13];
  const float* Wcb      = (const float*)d_in[14];
  const float* bcb      = (const float*)d_in[15];
  const float* Wf       = (const float*)d_in[16];
  const float* bf       = (const float*)d_in[17];
  float* out = (float*)d_out;

  _Float16* wW1 = (_Float16*)d_ws;          // 24576 halves
  _Float16* wW2 = wW1 + NW1;                // 24576 halves (96 KB total)

  cvt_weights_kernel<<<(NW1 + 255) / 256, 256, 0, stream>>>(W1, W2, wW1);

  feat_kernel<<<Bc * Tc, 256, 0, stream>>>(trend, seasonal, Wt, bt, Ws, bs, out);

  // 512 blocks x 8 waves x 16 samples = 65536 samples
  flow_kernel<<<(Bc * Tc) / 128, 256, 0, stream>>>(
      residual, b_init, Wc0, bc0, b1, b2, Wcb, bcb, Wf, bf, wW1, wW2, out);
}
